// EdgeConvBlock_1623497638706
// MI455X (gfx1250) — matrix operand order repaired and checked by reading, not compiled, not run on hardware
//
#include <hip/hip_runtime.h>
#include <hip/hip_bf16.h>

typedef _Float16 v16h __attribute__((ext_vector_type(16)));
typedef float    v8f  __attribute__((ext_vector_type(8)));
typedef _Float16 h2_t __attribute__((ext_vector_type(2)));

#define EPSV 1e-5f

// ---------------------------------------------------------------------------
// WMMA helper: D = A(16x32 f16) * B(32x16 f16) + C(16x16 f32)
// ---------------------------------------------------------------------------
__device__ __forceinline__ v8f wmma_f16(v16h a, v16h b, v8f c) {
    return __builtin_amdgcn_wmma_f32_16x16x32_f16(
        /*neg_a=*/false, a, /*neg_b=*/false, b,
        /*c_mod=*/(short)0, c, /*reuse_a=*/false, /*reuse_b=*/false);
}

// A-matrix 16-bit layout (ISA 7.12.2): lane l<16 -> row M=l, K in
// {0..7, 16..23}; lane>=16 -> row M=l-16, K in {8..15, 24..31}.
// VGPR pair j holds K = a_kbase(j,half)+{0,1}.
__device__ __forceinline__ int a_kbase(int j, int half, int s) {
    int base = (j < 4) ? (2 * j) : (16 + 2 * (j - 4));
    return base + 8 * half + 32 * s;
}

// B-matrix (32x16, KxN) 16-bit layout: lane l<16 -> col N=l, K in
// {0..7, 16..23}; lane>=16 -> col N=l-16, K in {8..15, 24..31}.
// Loaded from a row-major B^T table bt[N][64] so K,K+1 are contiguous.
__device__ __forceinline__ v16h load_bT(const _Float16* __restrict__ bt,
                                        int col, int half, int s) {
    const unsigned* row = (const unsigned*)(bt + (size_t)col * 64);
    v16h b;
#pragma unroll
    for (int j = 0; j < 8; ++j) {
        int kb = a_kbase(j, half, s);
        h2_t hv = __builtin_bit_cast(h2_t, row[kb >> 1]);
        b[2 * j]     = hv.x;
        b[2 * j + 1] = hv.y;
    }
    return b;
}

// ---------------------------------------------------------------------------
// Prep: f32 weights -> f16 B^T tables; fold BN into scale/bias.
// ---------------------------------------------------------------------------
__global__ __launch_bounds__(256) void k_prep(
    const float* __restrict__ w1, const float* __restrict__ cw,
    const float* __restrict__ w3,
    const float* __restrict__ g1, const float* __restrict__ b1,
    const float* __restrict__ m1, const float* __restrict__ v1,
    const float* __restrict__ g2, const float* __restrict__ b2,
    const float* __restrict__ m2, const float* __restrict__ v2,
    const float* __restrict__ g3, const float* __restrict__ b3,
    const float* __restrict__ m3, const float* __restrict__ v3,
    _Float16* __restrict__ w1t, _Float16* __restrict__ cwt,
    _Float16* __restrict__ w3t, float* __restrict__ cwxyz,
    float* __restrict__ bnc) {
    int tid = threadIdx.x;
    for (int i = tid; i < 64 * 64; i += 256) {
        int o = i >> 6, c = i & 63;
        w1t[i] = (_Float16)w1[c * 64 + o];        // w1 is [in][out]
        cwt[i] = (_Float16)cw[o * 67 + 3 + c];    // conv_w is [out][3+in]
        w3t[i] = (_Float16)w3[c * 64 + o];        // w3 is [in][out]
    }
    for (int i = tid; i < 64 * 3; i += 256)
        cwxyz[i] = cw[(i / 3) * 67 + (i % 3)];
    if (tid < 64) {
        float s;
        s = g1[tid] * rsqrtf(v1[tid] + EPSV); bnc[tid]       = s; bnc[ 64 + tid] = b1[tid] - m1[tid] * s;
        s = g2[tid] * rsqrtf(v2[tid] + EPSV); bnc[128 + tid] = s; bnc[192 + tid] = b2[tid] - m2[tid] * s;
        s = g3[tid] * rsqrtf(v3[tid] + EPSV); bnc[256 + tid] = s; bnc[320 + tid] = b3[tid] - m3[tid] * s;
    }
}

// ---------------------------------------------------------------------------
// h = relu(bn1(x @ w1)) stored as f16. One wave per 16-row tile.
// ---------------------------------------------------------------------------
__global__ __launch_bounds__(256) void k_linear1(
    const float* __restrict__ x, const _Float16* __restrict__ w1t,
    const float* __restrict__ bnc, _Float16* __restrict__ hout, int ntiles) {
    int wave = threadIdx.x >> 5;
    int tile = blockIdx.x * 8 + wave;
    if (tile >= ntiles) return;
    int lane = threadIdx.x & 31;
    int l = lane & 15, half = lane >> 4;
    const float* s1 = bnc;
    const float* t1 = bnc + 64;

    const float* xr = x + (size_t)(tile * 16 + l) * 64;   // A row = l
    v8f c[4] = {};
#pragma unroll
    for (int s = 0; s < 2; ++s) {
        v16h a;
#pragma unroll
        for (int j = 0; j < 8; ++j) {
            int kb = a_kbase(j, half, s);
            float2 f = *(const float2*)(xr + kb);
            a[2 * j]     = (_Float16)f.x;
            a[2 * j + 1] = (_Float16)f.y;
        }
#pragma unroll
        for (int nt = 0; nt < 4; ++nt) {
            v16h b = load_bT(w1t, nt * 16 + l, half, s);
            c[nt] = wmma_f16(a, b, c[nt]);
        }
    }
#pragma unroll
    for (int nt = 0; nt < 4; ++nt) {
        int col = nt * 16 + l;
        float sc = s1[col], bi = t1[col];
#pragma unroll
        for (int r = 0; r < 8; ++r) {
            int m = tile * 16 + r + 8 * half;             // C row layout
            float y = c[nt][r] * sc + bi;
            y = y > 0.f ? y : 0.f;
            hout[(size_t)m * 64 + col] = (_Float16)y;
        }
    }
}

// ---------------------------------------------------------------------------
// Fused: neighbor gather + conv1x1 (WMMA, xyz part folded into C-init) +
// bn2/relu + max over K + final linear3 + bn3 + residual + relu.
// 16 waves per block, one wave per point; wave 0 does the 16-point tail GEMM.
// ---------------------------------------------------------------------------
__global__ __launch_bounds__(512) void k_edge(
    const float* __restrict__ p, const float* __restrict__ x,
    const int* __restrict__ idx, const _Float16* __restrict__ h,
    const _Float16* __restrict__ cwt, const float* __restrict__ cwxyz,
    const _Float16* __restrict__ w3t, const float* __restrict__ bnc,
    float* __restrict__ out) {
    __shared__ float relS[16][16][3];   // [wave][neighbor][xyz]
    __shared__ float fS[16][64];        // pooled features for the 16 points

    int wave = threadIdx.x >> 5;
    int lane = threadIdx.x & 31;
    int l = lane & 15, half = lane >> 4;
    int point = blockIdx.x * 16 + wave;

    const float* s2 = bnc + 128; const float* t2 = bnc + 192;
    const float* s3 = bnc + 256; const float* t3 = bnc + 320;

    // per-lane neighbor (A-matrix row l); both half-waves use the same row
    int nb = idx[(size_t)point * 16 + l];
    if (half == 0) {
        float px = p[(size_t)point * 3 + 0];
        float py = p[(size_t)point * 3 + 1];
        float pz = p[(size_t)point * 3 + 2];
        relS[wave][l][0] = p[(size_t)nb * 3 + 0] - px;
        relS[wave][l][1] = p[(size_t)nb * 3 + 1] - py;
        relS[wave][l][2] = p[(size_t)nb * 3 + 2] - pz;
    }
    __syncthreads();

    // C init = xyz contribution (3-term FMA per element, C-matrix layout)
    v8f c[4];
#pragma unroll
    for (int nt = 0; nt < 4; ++nt) {
        int o = nt * 16 + l;
        float w0 = cwxyz[o * 3 + 0], w1v = cwxyz[o * 3 + 1], w2v = cwxyz[o * 3 + 2];
#pragma unroll
        for (int r = 0; r < 8; ++r) {
            int m = r + 8 * half;
            c[nt][r] = relS[wave][m][0] * w0 + relS[wave][m][1] * w1v
                     + relS[wave][m][2] * w2v;
        }
    }

    // gather neighbor features (f16, L2-resident) and run the channel GEMM
    const unsigned* hrow = (const unsigned*)(h + (size_t)nb * 64);
#pragma unroll
    for (int s = 0; s < 2; ++s) {
        v16h a;
#pragma unroll
        for (int j = 0; j < 8; ++j) {
            int kb = a_kbase(j, half, s);
            h2_t hv = __builtin_bit_cast(h2_t, hrow[kb >> 1]);
            a[2 * j]     = hv.x;
            a[2 * j + 1] = hv.y;
        }
#pragma unroll
        for (int nt = 0; nt < 4; ++nt) {
            v16h b = load_bT(cwt, nt * 16 + l, half, s);
            c[nt] = wmma_f16(a, b, c[nt]);
        }
    }

    // bn2 + relu + max over the 16 neighbors (rows)
#pragma unroll
    for (int nt = 0; nt < 4; ++nt) {
        int col = nt * 16 + l;
        float sc = s2[col], bi = t2[col];
        float mx = -3.402823466e38f;
#pragma unroll
        for (int r = 0; r < 8; ++r) {
            float y = c[nt][r] * sc + bi;
            y = y > 0.f ? y : 0.f;
            mx = mx > y ? mx : y;
        }
        float other = __shfl_xor(mx, 16, 32);   // combine the two half-waves
        mx = mx > other ? mx : other;
        if (half == 0) fS[wave][col] = mx;
    }
    __syncthreads();

    if (wave != 0) return;

    // tail GEMM for the block's 16 points: out = relu(bn3(f @ w3) + x)
    v8f d[4] = {};
#pragma unroll
    for (int s = 0; s < 2; ++s) {
        v16h a;
#pragma unroll
        for (int j = 0; j < 8; ++j) {
            int kb = a_kbase(j, half, s);
            a[2 * j]     = (_Float16)fS[l][kb];
            a[2 * j + 1] = (_Float16)fS[l][kb + 1];
        }
#pragma unroll
        for (int nt = 0; nt < 4; ++nt) {
            v16h b = load_bT(w3t, nt * 16 + l, half, s);
            d[nt] = wmma_f16(a, b, d[nt]);
        }
    }
#pragma unroll
    for (int nt = 0; nt < 4; ++nt) {
        int col = nt * 16 + l;
        float sc = s3[col], bi = t3[col];
#pragma unroll
        for (int r = 0; r < 8; ++r) {
            int m = blockIdx.x * 16 + r + 8 * half;
            float y = d[nt][r] * sc + bi + x[(size_t)m * 64 + col];
            y = y > 0.f ? y : 0.f;
            out[(size_t)m * 64 + col] = y;
        }
    }
}

// ---------------------------------------------------------------------------
// Host launcher
// ---------------------------------------------------------------------------
extern "C" void kernel_launch(void* const* d_in, const int* in_sizes, int n_in,
                              void* d_out, int out_size, void* d_ws, size_t ws_size,
                              hipStream_t stream) {
    const float* p   = (const float*)d_in[0];
    const float* x   = (const float*)d_in[1];
    const int*   idx = (const int*)d_in[2];
    const float* w1  = (const float*)d_in[3];
    const float* g1  = (const float*)d_in[4];
    const float* b1  = (const float*)d_in[5];
    const float* m1  = (const float*)d_in[6];
    const float* v1  = (const float*)d_in[7];
    const float* cw  = (const float*)d_in[8];
    const float* g2  = (const float*)d_in[9];
    const float* b2  = (const float*)d_in[10];
    const float* m2  = (const float*)d_in[11];
    const float* v2  = (const float*)d_in[12];
    const float* w3  = (const float*)d_in[13];
    const float* g3  = (const float*)d_in[14];
    const float* b3  = (const float*)d_in[15];
    const float* m3  = (const float*)d_in[16];
    const float* v3  = (const float*)d_in[17];
    float* out = (float*)d_out;

    const int n = in_sizes[0] / 3;          // number of points (100000)
    const int ntiles = n >> 4;              // n divisible by 16 here

    // workspace layout
    char* ws = (char*)d_ws;
    const size_t H_BYTES = (size_t)n * 64 * sizeof(_Float16);
    _Float16* h     = (_Float16*)ws;
    _Float16* w1t   = (_Float16*)(ws + H_BYTES);
    _Float16* cwt   = (_Float16*)(ws + H_BYTES + 8192);
    _Float16* w3t   = (_Float16*)(ws + H_BYTES + 16384);
    float*    cwxyz = (float*)   (ws + H_BYTES + 24576);
    float*    bnc   = (float*)   (ws + H_BYTES + 25600);

    k_prep<<<1, 256, 0, stream>>>(w1, cw, w3,
                                  g1, b1, m1, v1, g2, b2, m2, v2, g3, b3, m3, v3,
                                  w1t, cwt, w3t, cwxyz, bnc);

    int blocks1 = (ntiles + 7) / 8;
    k_linear1<<<blocks1, 256, 0, stream>>>(x, w1t, bnc, h, ntiles);

    k_edge<<<ntiles, 512, 0, stream>>>(p, x, idx, h, cwt, cwxyz, w3t, bnc, out);
}
